// ALISTA_18923625906623
// MI455X (gfx1250) — matrix operand order repaired and checked by reading, not compiled, not run on hardware
//
#include <hip/hip_runtime.h>
#include <math.h>

typedef __bf16 bf16;
typedef __attribute__((ext_vector_type(16))) __bf16 v16bf;
typedef __attribute__((ext_vector_type(8)))  __bf16 v8bf;
typedef __attribute__((ext_vector_type(8)))  float  v8f;

#define MM 256      // sensing dim m
#define NN 1024     // code dim n
#define BB 32768    // batch
#define KITER 16    // ALISTA iterations
#define ROWS 32     // batch rows per block (two WMMA M tiles -> B-frag reuse x2)

// bf16 <-> f32 bit helpers (for packed threshold path)
__device__ __forceinline__ float bf2f(unsigned int ubits) {
  union { unsigned int i; float f; } c; c.i = ubits << 16; return c.f;
}
__device__ __forceinline__ unsigned int f2bf(float f) {   // round-to-nearest-even
  union { float f; unsigned int i; } c; c.f = f;
  return (c.i + 0x7FFFu + ((c.i >> 16) & 1u)) >> 16;
}

// --- one-time input prep: A -> bf16 row-major, W -> bf16 transposed [NN][MM] ---
__global__ void cvt_A(const float* __restrict__ A, bf16* __restrict__ Abf) {
  int i = blockIdx.x * blockDim.x + threadIdx.x;
  if (i < MM * NN) Abf[i] = (bf16)A[i];
}
__global__ void cvt_Wt(const float* __restrict__ W, bf16* __restrict__ Wt) {
  int i = blockIdx.x * blockDim.x + threadIdx.x;
  if (i < MM * NN) {
    int m = i >> 10;          // row of W
    int n = i & (NN - 1);     // col of W
    Wt[n * MM + m] = (bf16)W[i];
  }
}

// --- main persistent-tile kernel: one block = 32 batch rows, all 16 iterations ---
__global__ __launch_bounds__(256)
void alista_kernel(const float* __restrict__ y,
                   const bf16*  __restrict__ Abf,   // [MM][NN] row-major bf16
                   const bf16*  __restrict__ Wt,    // [NN][MM] (W transposed) bf16
                   const float* __restrict__ step_sizes,
                   const float* __restrict__ thresholds,
                   float* __restrict__ out)
{
  extern __shared__ char smem[];
  bf16*           Xlds  = (bf16*)smem;                            // 32*1024 bf16 (64 KB)
  unsigned short* Glds  = (unsigned short*)(smem + 65536);        // 32*1024 bf16 (64 KB)
  bf16*           Rlds  = (bf16*)(smem + 131072);                 // 32*256  bf16 (16 KB)
  int*            redI  = (int*)(smem + 147456);                  // 32*8
  unsigned int*   rowLoI = (unsigned int*)(smem + 148480);        // 32
  unsigned int*   rowHiI = (unsigned int*)(smem + 148608);        // 32

  const int tid  = threadIdx.x;
  const int lane = tid & 31;
  const int wv   = tid >> 5;         // 8 waves
  const int hs   = lane >> 4;        // lane half-select (0/1)
  const int lm   = lane & 15;
  const int rowBase = blockIdx.x * ROWS;

  // zero x tile
  for (int c = tid; c < ROWS * NN; c += 256) Xlds[c] = (bf16)0.0f;
  __syncthreads();

  for (int k = 0; k < KITER; ++k) {
    const float gamma = fabsf(step_sizes[k]);
    const float theta = fabsf(thresholds[k]);
    const int   p     = (64 * (k + 1) < NN) ? 64 * (k + 1) : NN;

    // ---------- GEMM1: P = X (32x1024) * A^T (1024x256); R = y - P ----------
    // wave wv owns output cols [wv*32, wv*32+32): 2 col tiles x 2 row tiles.
    // One B-fragment (A rows) feeds TWO wmma (row tiles 0-15 and 16-31).
    #pragma unroll
    for (int t = 0; t < 2; ++t) {
      const int mt = wv * 32 + t * 16;
      v8f acc0 = {}, acc1 = {};
      const bf16* arow0 = Xlds + lm * NN;            // batch rows 0..15
      const bf16* arow1 = Xlds + (16 + lm) * NN;     // batch rows 16..31
      const bf16* brow  = Abf + (size_t)(mt + lm) * NN;
      #pragma unroll 4
      for (int n0 = 0; n0 < NN; n0 += 32) {
        __builtin_prefetch(brow + n0 + 128, 0, 1);   // global_prefetch_b8
        v16bf bfr, af0, af1;
        ((v8bf*)&bfr)[0] = *(const v8bf*)(brow + n0 + hs * 8);
        ((v8bf*)&bfr)[1] = *(const v8bf*)(brow + n0 + hs * 8 + 16);
        ((v8bf*)&af0)[0] = *(const v8bf*)(arow0 + n0 + hs * 8);
        ((v8bf*)&af0)[1] = *(const v8bf*)(arow0 + n0 + hs * 8 + 16);
        ((v8bf*)&af1)[0] = *(const v8bf*)(arow1 + n0 + hs * 8);
        ((v8bf*)&af1)[1] = *(const v8bf*)(arow1 + n0 + hs * 8 + 16);
        acc0 = __builtin_amdgcn_wmma_f32_16x16x32_bf16(false, af0, false, bfr,
                                                       (short)0, acc0, false, false);
        acc1 = __builtin_amdgcn_wmma_f32_16x16x32_bf16(false, af1, false, bfr,
                                                       (short)0, acc1, false, false);
      }
      const int col = mt + lm;
      #pragma unroll
      for (int r = 0; r < 8; ++r) {
        const int row0 = r + hs * 8;
        const float yv0 = y[(size_t)(rowBase + row0) * MM + col];
        Rlds[row0 * MM + col] = (bf16)(yv0 - acc0[r]);
      }
      #pragma unroll
      for (int r = 0; r < 8; ++r) {
        const int row1 = 16 + r + hs * 8;
        const float yv1 = y[(size_t)(rowBase + row1) * MM + col];
        Rlds[row1 * MM + col] = (bf16)(yv1 - acc1[r]);
      }
    }
    __syncthreads();

    // ---------- GEMM2: G = X + gamma * (R (32x256) * W (256x1024)) ----------
    // wave wv owns output cols [wv*128, wv*128+128): 8 col tiles x 2 row tiles.
    for (int t = 0; t < 8; ++t) {
      const int n0 = wv * 128 + t * 16;
      v8f acc0 = {}, acc1 = {};
      const bf16* arow0 = Rlds + lm * MM;
      const bf16* arow1 = Rlds + (16 + lm) * MM;
      const bf16* brow  = Wt + (size_t)(n0 + lm) * MM;
      #pragma unroll
      for (int k0 = 0; k0 < MM; k0 += 32) {
        __builtin_prefetch(brow + k0 + 64, 0, 1);
        v16bf bfr, af0, af1;
        ((v8bf*)&bfr)[0] = *(const v8bf*)(brow + k0 + hs * 8);
        ((v8bf*)&bfr)[1] = *(const v8bf*)(brow + k0 + hs * 8 + 16);
        ((v8bf*)&af0)[0] = *(const v8bf*)(arow0 + k0 + hs * 8);
        ((v8bf*)&af0)[1] = *(const v8bf*)(arow0 + k0 + hs * 8 + 16);
        ((v8bf*)&af1)[0] = *(const v8bf*)(arow1 + k0 + hs * 8);
        ((v8bf*)&af1)[1] = *(const v8bf*)(arow1 + k0 + hs * 8 + 16);
        acc0 = __builtin_amdgcn_wmma_f32_16x16x32_bf16(false, af0, false, bfr,
                                                       (short)0, acc0, false, false);
        acc1 = __builtin_amdgcn_wmma_f32_16x16x32_bf16(false, af1, false, bfr,
                                                       (short)0, acc1, false, false);
      }
      const int col = n0 + lm;
      #pragma unroll
      for (int r = 0; r < 8; ++r) {
        const int row0 = r + hs * 8;
        const float g0 = (float)Xlds[row0 * NN + col] + gamma * acc0[r];
        Glds[row0 * NN + col] = (unsigned short)f2bf(g0);
      }
      #pragma unroll
      for (int r = 0; r < 8; ++r) {
        const int row1 = 16 + r + hs * 8;
        const float g1 = (float)Xlds[row1 * NN + col] + gamma * acc1[r];
        Glds[row1 * NN + col] = (unsigned short)f2bf(g1);
      }
    }
    __syncthreads();

    // ---------- support-select soft threshold ----------
    // |bf16| bit patterns are order-isomorphic to magnitudes -> exact integer
    // bisection over [0, 2^15] (15 steps) for the p-th largest per row.
    // 8 threads per row; each owns 128 contiguous columns (64 packed dwords).
    const int trow = tid >> 3;
    const int seg  = tid & 7;
    const unsigned int* gw = (const unsigned int*)Glds + trow * (NN / 2) + seg * 64;
    unsigned int* xd = (unsigned int*)Xlds + trow * (NN / 2) + seg * 64;

    if (p < NN) {                       // p is block-uniform -> barriers are safe
      unsigned int mx = 0;
      for (int i = 0; i < 64; ++i) {
        const unsigned int v = gw[i] & 0x7FFF7FFFu;
        const unsigned int a = v & 0xFFFFu, b = v >> 16;
        mx = max(mx, max(a, b));
      }
      redI[trow * 8 + seg] = (int)mx;
      __syncthreads();
      if (seg == 0) {
        unsigned int m = 0;
        for (int j = 0; j < 8; ++j) m = max(m, (unsigned int)redI[trow * 8 + j]);
        rowHiI[trow] = m + 1u;
        rowLoI[trow] = 0u;
      }
      __syncthreads();
      for (int it = 0; it < 15; ++it) {
        const unsigned int mid = (rowLoI[trow] + rowHiI[trow]) >> 1;
        int cnt = 0;
        for (int i = 0; i < 64; ++i) {
          const unsigned int v = gw[i] & 0x7FFF7FFFu;
          cnt += ((v & 0xFFFFu) >= mid) ? 1 : 0;
          cnt += ((v >> 16) >= mid) ? 1 : 0;
        }
        redI[trow * 8 + seg] = cnt;
        __syncthreads();
        if (seg == 0) {
          int tot = 0;
          for (int j = 0; j < 8; ++j) tot += redI[trow * 8 + j];
          if (tot >= p) rowLoI[trow] = mid; else rowHiI[trow] = mid;
        }
        __syncthreads();
      }
      const unsigned int tbits = rowLoI[trow];   // p-th largest |g| (bf16 bits)
      for (int i = 0; i < 64; ++i) {
        const unsigned int v = gw[i];
        unsigned int r01 = 0;
        #pragma unroll
        for (int h = 0; h < 2; ++h) {
          const unsigned int u  = (v >> (16 * h)) & 0xFFFFu;
          const unsigned int az = u & 0x7FFFu;
          unsigned int res;
          if (az >= tbits) {
            res = u;                              // on support: keep
          } else {
            const float m = fmaxf(bf2f(az) - theta, 0.f);
            res = f2bf(m) | (u & 0x8000u);        // soft threshold, restore sign
          }
          r01 |= res << (16 * h);
        }
        xd[i] = r01;
      }
    } else {
      // p == N: support covers everything -> x = g (bf16 passthrough)
      for (int i = 0; i < 64; ++i) xd[i] = gw[i];
    }
    __syncthreads();
  }

  // write final x (coalesced)
  for (int c = tid; c < ROWS * NN; c += 256) {
    const int r = c >> 10, col = c & (NN - 1);
    out[(size_t)(rowBase + r) * NN + col] = (float)Xlds[c];
  }
}

extern "C" void kernel_launch(void* const* d_in, const int* in_sizes, int n_in,
                              void* d_out, int out_size, void* d_ws, size_t ws_size,
                              hipStream_t stream) {
  const float* y    = (const float*)d_in[0];
  const float* A    = (const float*)d_in[1];
  const float* W    = (const float*)d_in[2];
  const float* step = (const float*)d_in[3];
  const float* thr  = (const float*)d_in[4];

  bf16* Abf = (bf16*)d_ws;             // 256*1024 bf16 = 512 KB
  bf16* Wt  = Abf + MM * NN;           // 256*1024 bf16 = 512 KB (transposed)

  cvt_A <<<(MM * NN + 255) / 256, 256, 0, stream>>>(A, Abf);
  cvt_Wt<<<(MM * NN + 255) / 256, 256, 0, stream>>>(W, Wt);

  const size_t smem_bytes = 65536 + 65536 + 16384 + 1024 + 128 + 128; // 148736 B
  alista_kernel<<<BB / ROWS, 256, smem_bytes, stream>>>(y, Abf, Wt, step, thr,
                                                        (float*)d_out);
}
